// Encoder_47107201302764
// MI455X (gfx1250) — compile-verified, matrix operand order checked by reading, not run on hardware
//
#include <hip/hip_runtime.h>
#include <hip/hip_bf16.h>

// ---------------------------------------------------------------------------
// GraphConv encoder for MI455X (gfx1250, wave32, WMMA bf16 16x16x32).
//   h1 = relu([mean_agg(x) | x] @ [w1_rel;w1_root] + b1)     (K=256,  N=1024)
//   y2r2 = h1 @ [w2_rel | w2_root]                           (K=1024, N=1024)
//   h2 = relu(mean_agg(y2)*invdeg + r2 + b2)
//   P  = h2 @ [wmu_rel|wls_rel|wmu_root|wls_root]            (K=512,  N=32)
//   mu/logstd = mean_agg(P[:,:16])*invdeg + P[:,16:] + bias
// Mean-aggregation commutes with the linear maps -> aggregate the narrower
// side (128 / 512 / 16 dims). Weights are packed ONCE per launch into a
// transposed [N][K] bf16 layout so GEMM B-staging is a straight b128 copy
// (no in-kernel transpose VALU).
// ---------------------------------------------------------------------------

typedef __attribute__((ext_vector_type(16))) __bf16 v16bf;
typedef __attribute__((ext_vector_type(8)))  float  v8f;

#define BM 128
#define BN 128
#define BK 32
#define SROW 20   // u32 per LDS row: 16 data + 4 pad (80B: 16B-aligned, conflict-free)

static constexpr int NN  = 20000;
static constexpr int NE  = 160000;
static constexpr int FI  = 128;
static constexpr int H1S = 1024;
static constexpr int H2S = 512;

// ---------------- workspace layout (phase-overlapped) ----------------------
static constexpr size_t OFF_DEG  = 0;                                   // 20000 f32
static constexpr size_t OFF_H1   = 81920;                               // NN*1024 bf16
static constexpr size_t OFF_H2   = OFF_H1 + (size_t)NN * H1S * 2;       // NN*512 bf16
static constexpr size_t OFF_SCR  = OFF_H2 + (size_t)NN * H2S * 2;
// phase 1 scratch
static constexpr size_t OFF_AGG1 = OFF_SCR;                             // NN*128 f32
static constexpr size_t OFF_ACAT = OFF_AGG1 + (size_t)NN * FI * 4;      // NN*256 bf16
static constexpr size_t OFF_W1   = OFF_ACAT + (size_t)NN * 256 * 2;     // 1024*256 bf16 [N][K]
// phase 2 scratch
static constexpr size_t OFF_W2   = OFF_SCR;                             // 1024*1024 bf16 [N][K]
static constexpr size_t OFF_Y2   = OFF_W2 + (size_t)H1S * H1S * 2;      // NN*1024 f32
static constexpr size_t OFF_AGG2 = OFF_Y2 + (size_t)NN * H1S * 4;       // NN*512 f32
// phase 3 scratch
static constexpr size_t OFF_WH   = OFF_SCR;                             // 32*512 bf16 [N][K]
static constexpr size_t OFF_P    = OFF_WH + (size_t)H2S * 32 * 2;       // NN*32 f32
static constexpr size_t OFF_AGGP = OFF_P + (size_t)NN * 32 * 4;         // NN*16 f32

// ---------------------------------------------------------------------------
// LDS fragment load: rows hold 32 bf16 (16 u32) padded to SROW u32.
// 16x32 bf16 fragment (ISA 7.12.2): lane = M%16 (B: N%16), half = lane>>4,
// u32 p0..3 <- K/2 = half*4 + p ; p4..7 <- 8 + half*4 + (p-4).
// Both halves are contiguous 16B chunks -> two ds_load_b128.
// ---------------------------------------------------------------------------
__device__ __forceinline__ v16bf ld_frag(const unsigned int* s, int rbase, int lane) {
  const int r = rbase + (lane & 15);
  const int half = lane >> 4;
  const unsigned int* p = s + r * SROW + half * 4;
  union { uint4 q[2]; v16bf v; } f;
  f.q[0] = *(const uint4*)(p);
  f.q[1] = *(const uint4*)(p + 8);
  return f.v;
}

// ---------------------------------------------------------------------------
// Tiled WMMA GEMM: C[M,N] = A[M,K] @ Bt[N,K]^T. A row-major, Bt pre-transposed
// n-major (both bf16, K multiple of 32). 256 threads = 8 waves; wave grid
// 4(M) x 2(N); 32x64 per wave -> 8 accumulators, 8 WMMAs per K-tile.
// LDS double-buffered: one barrier per K-tile, next tile's global_load_b128s
// issued before the WMMAs. MODE 0: store f32. MODE 1: bias+relu, store bf16.
// ---------------------------------------------------------------------------
template <int MODE>
__global__ __launch_bounds__(256) void k_gemm(
    const __bf16* __restrict__ A, const __bf16* __restrict__ Bt,
    float* __restrict__ Cf, __bf16* __restrict__ Cb,
    const float* __restrict__ bias, int M, int N, int K) {
  __shared__ unsigned int sA[2][BM * SROW];  // 2 x 10240 B
  __shared__ unsigned int sB[2][BN * SROW];  // 2 x 10240 B

  const int tid  = threadIdx.x;
  const int lane = tid & 31;
  const int wave = tid >> 5;
  const int wm   = (wave & 3) * 32;
  const int wn   = (wave >> 2) * 64;
  const int bm0  = blockIdx.y * BM;
  const int bn0  = blockIdx.x * BN;

  // staging map: thread -> (row = tid>>1, half = tid&1): 16 bf16 = 2 x uint4
  const int row  = tid >> 1;
  const int half = tid & 1;
  const bool va  = (bm0 + row) < M;
  const bool vb  = (bn0 + row) < N;
  const __bf16* pa = A  + (size_t)(bm0 + row) * K + half * 16;
  const __bf16* pb = Bt + (size_t)(bn0 + row) * K + half * 16;
  const int soff = row * SROW + half * 8;

  const uint4 z4 = {0, 0, 0, 0};
  v8f zacc = {};
  v8f acc[2][4];
#pragma unroll
  for (int i = 0; i < 2; ++i)
#pragma unroll
    for (int j = 0; j < 4; ++j) acc[i][j] = zacc;

  // ---- prologue: stage K-tile 0 ----
  uint4 ra0 = z4, ra1 = z4, rb0 = z4, rb1 = z4;
  if (va) { const uint4* s = (const uint4*)pa; ra0 = s[0]; ra1 = s[1]; }
  if (vb) { const uint4* s = (const uint4*)pb; rb0 = s[0]; rb1 = s[1]; }
  {
    unsigned int* da = &sA[0][soff];
    unsigned int* db = &sB[0][soff];
    *(uint4*)(da) = ra0; *(uint4*)(da + 4) = ra1;
    *(uint4*)(db) = rb0; *(uint4*)(db + 4) = rb1;
  }
  __syncthreads();

  const int nt = K / BK;
  int cur = 0;
  for (int t = 0; t < nt; ++t) {
    // issue next tile's global loads early (hide under WMMA)
    uint4 na0 = z4, na1 = z4, nb0 = z4, nb1 = z4;
    if (t + 1 < nt) {
      const int kb = (t + 1) * BK;
      if (va) { const uint4* s = (const uint4*)(pa + kb); na0 = s[0]; na1 = s[1]; }
      if (vb) { const uint4* s = (const uint4*)(pb + kb); nb0 = s[0]; nb1 = s[1]; }
    }

    // ---- compute from buffer `cur`: 2x4 WMMA per wave ----
    const unsigned int* cA = sA[cur];
    const unsigned int* cB = sB[cur];
    v16bf a0 = ld_frag(cA, wm, lane);
    v16bf a1 = ld_frag(cA, wm + 16, lane);
    v16bf b0 = ld_frag(cB, wn, lane);
    v16bf b1 = ld_frag(cB, wn + 16, lane);
    v16bf b2 = ld_frag(cB, wn + 32, lane);
    v16bf b3 = ld_frag(cB, wn + 48, lane);
    acc[0][0] = __builtin_amdgcn_wmma_f32_16x16x32_bf16(false, a0, false, b0, (short)0, acc[0][0], false, false);
    acc[0][1] = __builtin_amdgcn_wmma_f32_16x16x32_bf16(false, a0, false, b1, (short)0, acc[0][1], false, false);
    acc[0][2] = __builtin_amdgcn_wmma_f32_16x16x32_bf16(false, a0, false, b2, (short)0, acc[0][2], false, false);
    acc[0][3] = __builtin_amdgcn_wmma_f32_16x16x32_bf16(false, a0, false, b3, (short)0, acc[0][3], false, false);
    acc[1][0] = __builtin_amdgcn_wmma_f32_16x16x32_bf16(false, a1, false, b0, (short)0, acc[1][0], false, false);
    acc[1][1] = __builtin_amdgcn_wmma_f32_16x16x32_bf16(false, a1, false, b1, (short)0, acc[1][1], false, false);
    acc[1][2] = __builtin_amdgcn_wmma_f32_16x16x32_bf16(false, a1, false, b2, (short)0, acc[1][2], false, false);
    acc[1][3] = __builtin_amdgcn_wmma_f32_16x16x32_bf16(false, a1, false, b3, (short)0, acc[1][3], false, false);

    if (t + 1 < nt) {
      // stage next tile into the other buffer; single barrier per K-tile
      unsigned int* da = &sA[cur ^ 1][soff];
      unsigned int* db = &sB[cur ^ 1][soff];
      *(uint4*)(da) = na0; *(uint4*)(da + 4) = na1;
      *(uint4*)(db) = nb0; *(uint4*)(db + 4) = nb1;
      __syncthreads();
      cur ^= 1;
    }
  }

  // ---- epilogue: C/D layout (ISA 7.12.2): VGPR r, lanes<16 -> M=r, else M=r+8
  const int chalf = lane >> 4;
  const int nl    = lane & 15;
#pragma unroll
  for (int fi = 0; fi < 2; ++fi) {
#pragma unroll
    for (int fj = 0; fj < 4; ++fj) {
      const int mb = bm0 + wm + fi * 16 + chalf * 8;
      const int n  = bn0 + wn + fj * 16 + nl;
      if (n < N) {
#pragma unroll
        for (int r = 0; r < 8; ++r) {
          const int m = mb + r;
          if (m < M) {
            float v = acc[fi][fj][r];
            if constexpr (MODE == 1) {
              v += bias[n];
              v = fmaxf(v, 0.0f);
              Cb[(size_t)m * N + n] = (__bf16)v;
            } else {
              Cf[(size_t)m * N + n] = v;
            }
          }
        }
      }
    }
  }
}

// ---------------------------- elementwise / graph kernels ------------------
__global__ void k_zero(float* __restrict__ p, int n) {
  int i = blockIdx.x * blockDim.x + threadIdx.x;
  if (i < n) p[i] = 0.0f;
}

__global__ void k_deg(const long long* __restrict__ eidx, float* __restrict__ deg, int E) {
  int e = blockIdx.x * blockDim.x + threadIdx.x;
  if (e < E) atomicAdd(&deg[(int)eidx[E + e]], 1.0f);
}

__global__ void k_invdeg(float* __restrict__ deg, int n) {
  int i = blockIdx.x * blockDim.x + threadIdx.x;
  if (i < n) deg[i] = 1.0f / fmaxf(deg[i], 1.0f);
}

// scatter-add x[src] rows into agg[dst]; 4 floats per thread (float4 gather)
__global__ void k_scatter_add(const float* __restrict__ x, int ldx,
                              const long long* __restrict__ eidx, int E,
                              float* __restrict__ agg, int ldagg, int Fq) {
  int idx = blockIdx.x * blockDim.x + threadIdx.x;
  if (idx >= E * Fq) return;
  int e = idx / Fq;
  int c = (idx - e * Fq) * 4;
  int s = (int)eidx[e];
  int d = (int)eidx[E + e];
  float4 v = *(const float4*)(x + (size_t)s * ldx + c);
  float* a = agg + (size_t)d * ldagg + c;
  atomicAdd(a + 0, v.x);
  atomicAdd(a + 1, v.y);
  atomicAdd(a + 2, v.z);
  atomicAdd(a + 3, v.w);
}

// a_cat[n, 0:128] = agg1*invdeg ; a_cat[n, 128:256] = x   (bf16)
__global__ void k_cat1(const float* __restrict__ agg1, const float* __restrict__ inv,
                       const float* __restrict__ x, __bf16* __restrict__ acat) {
  int idx = blockIdx.x * blockDim.x + threadIdx.x;
  if (idx >= NN * FI) return;
  int n = idx >> 7, f = idx & 127;
  acat[(size_t)n * 256 + f]       = (__bf16)(agg1[idx] * inv[n]);
  acat[(size_t)n * 256 + 128 + f] = (__bf16)x[idx];
}

// ---- one-time weight packs: f32 [K,N] -> bf16 transposed [N,K] ------------
// w1t[n*256 + k]: k<128 from w1_rel[k,n], else w1_root[k-128,n]   (n<1024)
__global__ void k_packw1_t(const float* __restrict__ w1rel, const float* __restrict__ w1rt,
                           __bf16* __restrict__ out) {
  int idx = blockIdx.x * blockDim.x + threadIdx.x;
  if (idx >= H1S * 256) return;
  int n = idx >> 8, k = idx & 255;
  float v = (k < 128) ? w1rel[(size_t)k * H1S + n] : w1rt[(size_t)(k - 128) * H1S + n];
  out[idx] = (__bf16)v;
}

// w2t[n*1024 + k]: n<512 -> w2_rel[k,n]; n>=512 -> w2_root[k,n-512]  (k<1024)
__global__ void k_packw2_t(const float* __restrict__ w2rel, const float* __restrict__ w2rt,
                           __bf16* __restrict__ out) {
  int idx = blockIdx.x * blockDim.x + threadIdx.x;
  if (idx >= H1S * 1024) return;
  int n = idx >> 10, k = idx & 1023;
  float v = (n < 512) ? w2rel[(size_t)k * 512 + n] : w2rt[(size_t)k * 512 + (n - 512)];
  out[idx] = (__bf16)v;
}

// wht[n*512 + k]: n 0:8|8:16|16:24|24:32 = wmu_rel|wls_rel|wmu_root|wls_root
__global__ void k_packheads_t(const float* __restrict__ a, const float* __restrict__ b,
                              const float* __restrict__ c, const float* __restrict__ d,
                              __bf16* __restrict__ out) {
  int idx = blockIdx.x * blockDim.x + threadIdx.x;
  if (idx >= 32 * H2S) return;
  int n = idx >> 9, k = idx & 511;
  int sel = n >> 3, col = n & 7;
  const float* src = (sel == 0) ? a : (sel == 1) ? b : (sel == 2) ? c : d;
  out[idx] = (__bf16)src[(size_t)k * 8 + col];
}

// h2 = relu(agg2*invdeg + root_part + b2)  (bf16)
__global__ void k_finish2(const float* __restrict__ agg2, const float* __restrict__ inv,
                          const float* __restrict__ y2r2, const float* __restrict__ b2,
                          __bf16* __restrict__ h2) {
  int idx = blockIdx.x * blockDim.x + threadIdx.x;
  if (idx >= NN * H2S) return;
  int n = idx >> 9, j = idx & 511;
  float v = agg2[idx] * inv[n] + y2r2[(size_t)n * 1024 + 512 + j] + b2[j];
  h2[idx] = (__bf16)fmaxf(v, 0.0f);
}

// mu / logstd heads
__global__ void k_finish3(const float* __restrict__ aggP, const float* __restrict__ inv,
                          const float* __restrict__ P, const float* __restrict__ bmu,
                          const float* __restrict__ bls, float* __restrict__ out) {
  int idx = blockIdx.x * blockDim.x + threadIdx.x;
  if (idx >= NN * 8) return;
  int n = idx >> 3, j = idx & 7;
  float mu = aggP[(size_t)n * 16 + j] * inv[n] + P[(size_t)n * 32 + 16 + j] + bmu[j];
  float ls = aggP[(size_t)n * 16 + 8 + j] * inv[n] + P[(size_t)n * 32 + 24 + j] + bls[j];
  out[idx] = mu;
  out[(size_t)NN * 8 + idx] = ls;
}

// ---------------------------------------------------------------------------
static inline int cdiv(int a, int b) { return (a + b - 1) / b; }

extern "C" void kernel_launch(void* const* d_in, const int* in_sizes, int n_in,
                              void* d_out, int out_size, void* d_ws, size_t ws_size,
                              hipStream_t stream) {
  const float*     x     = (const float*)d_in[0];
  const long long* eidx  = (const long long*)d_in[1];
  const float*     w1rel = (const float*)d_in[2];
  const float*     b1    = (const float*)d_in[3];
  const float*     w1rt  = (const float*)d_in[4];
  const float*     w2rel = (const float*)d_in[5];
  const float*     b2    = (const float*)d_in[6];
  const float*     w2rt  = (const float*)d_in[7];
  const float*     wmur  = (const float*)d_in[8];
  const float*     bmu   = (const float*)d_in[9];
  const float*     wmurt = (const float*)d_in[10];
  const float*     wlsr  = (const float*)d_in[11];
  const float*     bls   = (const float*)d_in[12];
  const float*     wlsrt = (const float*)d_in[13];

  char* ws = (char*)d_ws;
  float*  deg  = (float*)(ws + OFF_DEG);
  __bf16* h1   = (__bf16*)(ws + OFF_H1);
  __bf16* h2   = (__bf16*)(ws + OFF_H2);
  float*  agg1 = (float*)(ws + OFF_AGG1);
  __bf16* acat = (__bf16*)(ws + OFF_ACAT);
  __bf16* w1t  = (__bf16*)(ws + OFF_W1);
  __bf16* w2t  = (__bf16*)(ws + OFF_W2);
  float*  y2r2 = (float*)(ws + OFF_Y2);
  float*  agg2 = (float*)(ws + OFF_AGG2);
  __bf16* wht  = (__bf16*)(ws + OFF_WH);
  float*  P    = (float*)(ws + OFF_P);
  float*  aggP = (float*)(ws + OFF_AGGP);
  float*  out  = (float*)d_out;

  const int E = NE, T = 256;

  // degrees -> inverse degrees
  k_zero<<<cdiv(NN, T), T, 0, stream>>>(deg, NN);
  k_deg<<<cdiv(E, T), T, 0, stream>>>(eidx, deg, E);
  k_invdeg<<<cdiv(NN, T), T, 0, stream>>>(deg, NN);

  // ---- layer 1: aggregate x (128), fused [mean|x] @ [w1_rel;w1_root] ----
  k_zero<<<cdiv(NN * FI, T), T, 0, stream>>>(agg1, NN * FI);
  k_scatter_add<<<cdiv(E * (FI / 4), T), T, 0, stream>>>(x, FI, eidx, E, agg1, FI, FI / 4);
  k_cat1<<<cdiv(NN * FI, T), T, 0, stream>>>(agg1, deg, x, acat);
  k_packw1_t<<<cdiv(H1S * 256, T), T, 0, stream>>>(w1rel, w1rt, w1t);
  {
    dim3 g(cdiv(H1S, BN), cdiv(NN, BM));
    k_gemm<1><<<g, T, 0, stream>>>(acat, w1t, nullptr, h1, b1, NN, H1S, 256);
  }

  // ---- layer 2: project first (y2|r2), then aggregate 512-dim y2 ----
  k_packw2_t<<<cdiv(H1S * 1024, T), T, 0, stream>>>(w2rel, w2rt, w2t);
  {
    dim3 g(cdiv(1024, BN), cdiv(NN, BM));
    k_gemm<0><<<g, T, 0, stream>>>(h1, w2t, y2r2, nullptr, nullptr, NN, 1024, H1S);
  }
  k_zero<<<cdiv(NN * H2S, T), T, 0, stream>>>(agg2, NN * H2S);
  k_scatter_add<<<cdiv(E * (H2S / 4), T), T, 0, stream>>>(y2r2, 1024, eidx, E, agg2, H2S, H2S / 4);
  k_finish2<<<cdiv(NN * H2S, T), T, 0, stream>>>(agg2, deg, y2r2, b2, h2);

  // ---- heads: project to 32 cols, aggregate 16, combine ----
  k_packheads_t<<<cdiv(32 * H2S, T), T, 0, stream>>>(wmur, wlsr, wmurt, wlsrt, wht);
  {
    dim3 g(cdiv(32, BN), cdiv(NN, BM));
    k_gemm<0><<<g, T, 0, stream>>>(h2, wht, P, nullptr, nullptr, NN, 32, H2S);
  }
  k_zero<<<cdiv(NN * 16, T), T, 0, stream>>>(aggP, NN * 16);
  k_scatter_add<<<cdiv(E * 4, T), T, 0, stream>>>(P, 32, eidx, E, aggP, 16, 4);
  k_finish3<<<cdiv(NN * 8, T), T, 0, stream>>>(aggP, deg, P, bmu, bls, out);
}
